// Atom37Encoder_7627861918028
// MI455X (gfx1250) — matrix operand order repaired and checked by reading, not compiled, not run on hardware
//
#include <hip/hip_runtime.h>
#include <hip/hip_bf16.h>

typedef __attribute__((ext_vector_type(16))) __bf16 v16bf;
typedef __attribute__((ext_vector_type(8)))  float  v8f;

#define ETOT 30720
#define NTOT 1024

__device__ __forceinline__ v8f vzero() {
  v8f z = {0.f,0.f,0.f,0.f,0.f,0.f,0.f,0.f};
  return z;
}

__device__ __forceinline__ void store_bf4(__bf16* d, float a, float b, float c, float e) {
  union { __bf16 h[4]; uint2 u; } t;
  t.h[0] = (__bf16)a; t.h[1] = (__bf16)b; t.h[2] = (__bf16)c; t.h[3] = (__bf16)e;
  *(uint2*)d = t.u;
}

// A fragment (16xK row-major bf16), K-step 32.
__device__ __forceinline__ v16bf frag_a(const __bf16* base, int lda, int lane) {
  int m  = lane & 15;
  int k0 = (lane >> 4) << 3;
  const __bf16* p = base + m * lda + k0;
  v16bf r;
  ((uint4*)&r)[0] = *(const uint4*)(p);
  ((uint4*)&r)[1] = *(const uint4*)(p + 16);
  return r;
}

// B fragment from transposed layout Bt[n][k] (ldb elems).
__device__ __forceinline__ v16bf frag_bt(const __bf16* baseT, int ldb, int lane) {
  int n  = lane & 15;
  int k0 = (lane >> 4) << 3;
  const __bf16* p = baseT + n * ldb + k0;
  v16bf r;
  ((uint4*)&r)[0] = *(const uint4*)(p);
  ((uint4*)&r)[1] = *(const uint4*)(p + 16);
  return r;
}

__device__ __forceinline__ v8f wmma_bf(v16bf a, v16bf b, v8f c) {
  return __builtin_amdgcn_wmma_f32_16x16x32_bf16(false, a, false, b, (short)0, c, false, false);
}

// ---------------------------------------------------------------- utilities
__global__ __launch_bounds__(256) void k_zero(float* __restrict__ p, int n) {
  int i = blockIdx.x * 256 + threadIdx.x;
  if (i < n) p[i] = 0.f;
}

__global__ __launch_bounds__(256) void k_cnt(const int* __restrict__ eidx, float* __restrict__ cnt) {
  int i = blockIdx.x * 256 + threadIdx.x;
  if (i < ETOT) atomicAdd(&cnt[eidx[ETOT + i]], 1.f);
}

// transpose + convert: src f32 [K][N] -> dst bf16 [N][Kpad], zero pad k>=K
__global__ __launch_bounds__(256) void k_transpose(const float* __restrict__ src,
    __bf16* __restrict__ dst, int K, int N, int Kpad) {
  int idx = blockIdx.x * 256 + threadIdx.x;
  if (idx >= N * Kpad) return;
  int n = idx / Kpad, k = idx % Kpad;
  dst[idx] = (__bf16)((k < K) ? src[(size_t)k * N + n] : 0.f);
}

// ------------------------------------------------------------- node embed
__global__ __launch_bounds__(256) void k_node_embed(const float* __restrict__ nraw,
    const float* __restrict__ nws, const float* __restrict__ nwv,
    float* __restrict__ xs, float* __restrict__ xv) {
  int idx = blockIdx.x * 256 + threadIdx.x;
  if (idx >= NTOT * 56) return;
  int n = idx / 56, o = idx % 56;
  const float* row = nraw + (size_t)n * 139;
  if (o < 32) {
    float s = 0.f;
    #pragma unroll
    for (int i = 0; i < 28; ++i) s += row[i] * nws[i * 32 + o];
    xs[n * 32 + o] = s;
  } else {
    int q = o - 32, j = q / 3, x = q % 3;
    float s = 0.f;
    #pragma unroll
    for (int i = 0; i < 37; ++i) s += row[28 + i * 3 + x] * nwv[i * 8 + j];
    xv[n * 24 + j * 3 + x] = s;
  }
}

// ------------------------------------------------------------- edge embed
// 32 edges/block, 8 waves; dual M-tiles per wave share every B fragment;
// GEMM1 software-pipelines the B stream one step ahead.
#define K1_SMEM (106496 + 8192 + 8192 + 16384 + 256)
__global__ __launch_bounds__(256) void k_edge_embed(const float* __restrict__ eraw,
    const __bf16* __restrict__ w1t, const float* __restrict__ b1,
    const __bf16* __restrict__ w2t, const float* __restrict__ b2,
    const __bf16* __restrict__ w3t, const float* __restrict__ b3,
    const float* __restrict__ lng, const float* __restrict__ lnb,
    float* __restrict__ ef, __bf16* __restrict__ efb) {
  extern __shared__ char smem1[];
  __bf16* sA  = (__bf16*)smem1;                 // [32][1664]
  __bf16* sH1 = (__bf16*)(smem1 + 106496);      // [32][128]
  __bf16* sH2 = (__bf16*)(smem1 + 114688);      // [32][128]
  float*  sF  = (float*)(smem1 + 122880);       // [32][128]
  float*  sMV = (float*)(smem1 + 139264);       // [32][2]
  int tid = threadIdx.x, lane = tid & 31, wv = tid >> 5;
  int e0 = blockIdx.x * 32;
  int nl = (lane & 15) + wv * 16, mb = (lane >> 4) * 8;

  // stage full 32x1664 A tile, float4 -> bf16x4
  #pragma unroll 8
  for (int it = 0; it < 52; ++it) {
    int v4 = it * 256 + tid;
    int r = v4 / 416, c4 = (v4 % 416) * 4;
    const float4 f = *(const float4*)(eraw + (size_t)(e0 + r) * 1664 + c4);
    store_bf4(sA + r * 1664 + c4, f.x, f.y, f.z, f.w);
  }
  __syncthreads();

  // GEMM1 with one-step B prefetch
  v8f acc1a = vzero(), acc1b = vzero();
  const __bf16* bt1 = w1t + (size_t)(wv * 16) * 1664;
  v16bf b = frag_bt(bt1, 1664, lane);
  #pragma unroll 4
  for (int kk = 0; kk < 1664; kk += 32) {
    int kn = (kk + 32 < 1664) ? kk + 32 : kk;
    v16bf bn = frag_bt(bt1 + kn, 1664, lane);
    acc1a = wmma_bf(frag_a(sA + kk, 1664, lane), b, acc1a);
    acc1b = wmma_bf(frag_a(sA + 16 * 1664 + kk, 1664, lane), b, acc1b);
    b = bn;
  }
  {
    float bb = b1[nl];
    #pragma unroll
    for (int r = 0; r < 8; ++r) {
      float va = acc1a[r] + bb, vb = acc1b[r] + bb;
      sH1[(mb + r) * 128 + nl]      = (__bf16)(va > 0.f ? va : 0.f);
      sH1[(16 + mb + r) * 128 + nl] = (__bf16)(vb > 0.f ? vb : 0.f);
    }
  }
  __syncthreads();
  // GEMM2
  v8f acc2a = vzero(), acc2b = vzero();
  const __bf16* bt2 = w2t + (size_t)(wv * 16) * 128;
  #pragma unroll
  for (int ks = 0; ks < 4; ++ks) {
    v16bf bf = frag_bt(bt2 + ks * 32, 128, lane);
    acc2a = wmma_bf(frag_a(sH1 + ks * 32, 128, lane), bf, acc2a);
    acc2b = wmma_bf(frag_a(sH1 + 16 * 128 + ks * 32, 128, lane), bf, acc2b);
  }
  {
    float bb = b2[nl];
    #pragma unroll
    for (int r = 0; r < 8; ++r) {
      float va = acc2a[r] + bb, vb = acc2b[r] + bb;
      sH2[(mb + r) * 128 + nl]      = (__bf16)(va > 0.f ? va : 0.f);
      sH2[(16 + mb + r) * 128 + nl] = (__bf16)(vb > 0.f ? vb : 0.f);
    }
  }
  __syncthreads();
  // GEMM3
  v8f acc3a = vzero(), acc3b = vzero();
  const __bf16* bt3 = w3t + (size_t)(wv * 16) * 128;
  #pragma unroll
  for (int ks = 0; ks < 4; ++ks) {
    v16bf bf = frag_bt(bt3 + ks * 32, 128, lane);
    acc3a = wmma_bf(frag_a(sH2 + ks * 32, 128, lane), bf, acc3a);
    acc3b = wmma_bf(frag_a(sH2 + 16 * 128 + ks * 32, 128, lane), bf, acc3b);
  }
  {
    float bb = b3[nl];
    #pragma unroll
    for (int r = 0; r < 8; ++r) {
      sF[(mb + r) * 128 + nl]      = acc3a[r] + bb;
      sF[(16 + mb + r) * 128 + nl] = acc3b[r] + bb;
    }
  }
  __syncthreads();
  if (tid < 32) {
    float m = 0.f;
    for (int c = 0; c < 128; ++c) m += sF[tid * 128 + c];
    m *= (1.f / 128.f);
    float v = 0.f;
    for (int c = 0; c < 128; ++c) { float d = sF[tid * 128 + c] - m; v += d * d; }
    sMV[tid * 2] = m;
    sMV[tid * 2 + 1] = rsqrtf(v * (1.f / 128.f) + 1e-5f);
  }
  __syncthreads();
  #pragma unroll
  for (int it = 0; it < 16; ++it) {
    int i = it * 256 + tid;
    int r = i >> 7, c = i & 127;
    float y = (sF[i] - sMV[r * 2]) * sMV[r * 2 + 1] * lng[c] + lnb[c];
    ef[(size_t)(e0 + r) * 128 + c]  = y;
    efb[(size_t)(e0 + r) * 128 + c] = (__bf16)y;
  }
}

// -------------------------------------------- fused fc + tensor product + scatter
// 32 edges/block, dual M-tiles per wave (halves the dominant fc_w2 L2 stream);
// TP processes 2 edges per thread. ~237 KB LDS of CDNA5's 320 KB/WGP.
#define K2_SMEM (212992 + 8192 + 8192 + 4096 + 3072 + 384)
__global__ __launch_bounds__(256) void k_fc_tp(const __bf16* __restrict__ efb,
    const float* __restrict__ xs, const float* __restrict__ xv,
    const float* __restrict__ ev, const int* __restrict__ eidx,
    const __bf16* __restrict__ w1t, const float* __restrict__ b1,
    const __bf16* __restrict__ w2t, const float* __restrict__ b2,
    float* __restrict__ ags, float* __restrict__ agv) {
  extern __shared__ char smem2[];
  float*  sW  = (float*)smem2;                    // [32][1664]
  __bf16* sA  = (__bf16*)(smem2 + 212992);        // [32][128]
  __bf16* sHb = (__bf16*)(smem2 + 221184);        // [32][128]
  float*  sXS = (float*)(smem2 + 229376);         // [32][32]
  float*  sXV = (float*)(smem2 + 233472);         // [32][24]
  float*  sSH = (float*)(smem2 + 236544);         // [32][3]
  int tid = threadIdx.x, lane = tid & 31, wv = tid >> 5;
  int e0 = blockIdx.x * 32;
  int nl = (lane & 15) + wv * 16, mb = (lane >> 4) * 8;
  const int* dstp = eidx;

  // stage ef tile (4096 bf16 = 512 uint4) + node gathers
  ((uint4*)sA)[tid]       = ((const uint4*)(efb + (size_t)e0 * 128))[tid];
  ((uint4*)sA)[tid + 256] = ((const uint4*)(efb + (size_t)e0 * 128))[tid + 256];
  {                         // xs gather: 32 rows x 8 float4 = 256
    int r = tid >> 3, q = (tid & 7) * 4;
    ((float4*)sXS)[tid] = *(const float4*)(xs + (size_t)dstp[e0 + r] * 32 + q);
  }
  if (tid < 192) {          // xv gather: 32 rows x 6 float4
    int r = tid / 6, q = (tid % 6) * 4;
    ((float4*)sXV)[tid] = *(const float4*)(xv + (size_t)dstp[e0 + r] * 24 + q);
  } else if (tid < 224) {   // spherical harmonics (32 rows)
    int r = tid - 192;
    float vx = ev[(size_t)(e0 + r) * 3], vy = ev[(size_t)(e0 + r) * 3 + 1],
          vz = ev[(size_t)(e0 + r) * 3 + 2];
    float s = 1.7320508075688772f / (sqrtf(vx * vx + vy * vy + vz * vz) + 1e-8f);
    sSH[r * 3] = vx * s; sSH[r * 3 + 1] = vy * s; sSH[r * 3 + 2] = vz * s;
  }
  __syncthreads();

  // GEMM1: h = relu(ef @ fc_w1 + b1), dual M
  v8f acc1a = vzero(), acc1b = vzero();
  const __bf16* btw1 = w1t + (size_t)(wv * 16) * 128;
  #pragma unroll
  for (int ks = 0; ks < 4; ++ks) {
    v16bf bf = frag_bt(btw1 + ks * 32, 128, lane);
    acc1a = wmma_bf(frag_a(sA + ks * 32, 128, lane), bf, acc1a);
    acc1b = wmma_bf(frag_a(sA + 16 * 128 + ks * 32, 128, lane), bf, acc1b);
  }
  {
    float bb = b1[nl];
    #pragma unroll
    for (int r = 0; r < 8; ++r) {
      float va = acc1a[r] + bb, vb = acc1b[r] + bb;
      sHb[(mb + r) * 128 + nl]      = (__bf16)(va > 0.f ? va : 0.f);
      sHb[(16 + mb + r) * 128 + nl] = (__bf16)(vb > 0.f ? vb : 0.f);
    }
  }
  __syncthreads();

  // GEMM2: w = h @ fc_w2 + b2; both M A-frags hoisted; B shared, direct global
  v16bf af[4], af2[4];
  #pragma unroll
  for (int ks = 0; ks < 4; ++ks) {
    af[ks]  = frag_a(sHb + ks * 32, 128, lane);
    af2[ks] = frag_a(sHb + 16 * 128 + ks * 32, 128, lane);
  }
  for (int c = 0; c < 13; ++c) {
    v8f acc2a = vzero(), acc2b = vzero();
    const __bf16* btw2 = w2t + (size_t)(c * 128 + wv * 16) * 128;
    #pragma unroll
    for (int ks = 0; ks < 4; ++ks) {
      v16bf bf = frag_bt(btw2 + ks * 32, 128, lane);
      acc2a = wmma_bf(af[ks],  bf, acc2a);
      acc2b = wmma_bf(af2[ks], bf, acc2b);
    }
    int col = c * 128 + nl;
    float bb = b2[col];
    #pragma unroll
    for (int r = 0; r < 8; ++r) {
      sW[(mb + r) * 1664 + col]      = acc2a[r] + bb;
      sW[(16 + mb + r) * 1664 + col] = acc2b[r] + bb;
    }
  }
  __syncthreads();

  // per-edge tensor product; 16 threads per edge, 2 edges per thread
  {
    int rbase = tid >> 4, j = tid & 15;
    const float a1  = 0.125f;                  // 1/sqrt(64)
    const float a4s = 0.14433756729740646f;    // (1/4)/sqrt(3)
    const float a2  = 0.10206207261596575f;    // 1/sqrt(96)
    const float a3  = 0.20412414523193150f;    // 1/sqrt(24)
    const float a5s = 0.14433756729740646f;    // 1/sqrt(48)
    #pragma unroll
    for (int h2 = 0; h2 < 2; ++h2) {
      int r = rbase + h2 * 16;
      const float* W  = sW  + r * 1664;
      const float* XS = sXS + r * 32;
      const float* XV = sXV + r * 24;
      float shx = sSH[r * 3], shy = sSH[r * 3 + 1], shz = sSH[r * 3 + 2];
      int srcn = eidx[ETOT + e0 + r];
      float vd[8];
      #pragma unroll
      for (int i = 0; i < 8; ++i)
        vd[i] = XV[i * 3] * shx + XV[i * 3 + 1] * shy + XV[i * 3 + 2] * shz;
      #pragma unroll
      for (int h = 0; h < 2; ++h) {
        int jj = j + h * 16;
        float s1 = 0.f, s4 = 0.f;
        #pragma unroll
        for (int i = 0; i < 32; ++i) s1 += XS[i] * W[i * 32 + jj];
        #pragma unroll
        for (int i = 0; i < 8;  ++i) s4 += vd[i] * W[1344 + i * 32 + jj];
        atomicAdd(&ags[(size_t)srcn * 32 + jj], a1 * s1 + a4s * s4);
      }
      if (j < 8) {
        float t2 = 0.f;
        #pragma unroll
        for (int i = 0; i < 32; ++i) t2 += XS[i] * W[1024 + i * 8 + j];
        float cr[8][3];
        #pragma unroll
        for (int i = 0; i < 8; ++i) {
          float ax = XV[i * 3], ay = XV[i * 3 + 1], az = XV[i * 3 + 2];
          cr[i][0] = ay * shz - az * shy;
          cr[i][1] = az * shx - ax * shz;
          cr[i][2] = ax * shy - ay * shx;
        }
        float sh3[3] = {shx, shy, shz};
        #pragma unroll
        for (int x = 0; x < 3; ++x) {
          float t3 = 0.f, t5 = 0.f;
          #pragma unroll
          for (int i = 0; i < 8; ++i) {
            t3 += XV[i * 3 + x] * W[1280 + i * 8 + j];
            t5 += cr[i][x]      * W[1600 + i * 8 + j];
          }
          atomicAdd(&agv[(size_t)srcn * 24 + j * 3 + x],
                    a2 * t2 * sh3[x] + a3 * t3 + a5s * t5);
        }
      }
    }
  }
}

// ------------------------------------------------------- node update + BN stats
__global__ __launch_bounds__(256) void k_node_s(float* __restrict__ xs,
    const float* __restrict__ ags, const float* __restrict__ cnt,
    float* __restrict__ bnsum, float* __restrict__ bnsq) {
  __shared__ float sS[32], sQ[32];
  int tid = threadIdx.x;
  if (tid < 32) { sS[tid] = 0.f; sQ[tid] = 0.f; }
  __syncthreads();
  int idx = blockIdx.x * 256 + tid;
  int n = idx >> 5, c = idx & 31;
  float x = xs[idx] + ags[idx] / fmaxf(cnt[n], 1.f);
  xs[idx] = x;
  atomicAdd(&sS[c], x);
  atomicAdd(&sQ[c], x * x);
  __syncthreads();
  if (tid < 32) { atomicAdd(&bnsum[tid], sS[tid]); atomicAdd(&bnsq[tid], sQ[tid]); }
}

__global__ __launch_bounds__(256) void k_node_v(float* __restrict__ xv,
    const float* __restrict__ agv, const float* __restrict__ cnt,
    float* __restrict__ fnsum) {
  __shared__ float sF[8];
  int tid = threadIdx.x;
  if (tid < 8) sF[tid] = 0.f;
  __syncthreads();
  int idx = blockIdx.x * 256 + tid;
  int n = idx >> 3, j = idx & 7;
  float inv = 1.f / fmaxf(cnt[n], 1.f);
  float acc = 0.f;
  #pragma unroll
  for (int x = 0; x < 3; ++x) {
    int o = n * 24 + j * 3 + x;
    float t = xv[o] + agv[o] * inv;
    xv[o] = t;
    acc += t * t;
  }
  atomicAdd(&sF[j], acc);
  __syncthreads();
  if (tid < 8) atomicAdd(&fnsum[tid], sF[tid]);
}

__global__ __launch_bounds__(256) void k_bn_apply(float* __restrict__ xs, float* __restrict__ xv,
    __bf16* __restrict__ xsb, const float* __restrict__ g, const float* __restrict__ b,
    const float* __restrict__ vg, const float* __restrict__ bnsum,
    const float* __restrict__ bnsq, const float* __restrict__ fnsum) {
  int idx = blockIdx.x * 256 + threadIdx.x;
  if (idx < NTOT * 32) {
    int c = idx & 31;
    float m = bnsum[c] * (1.f / NTOT);
    float var = bnsq[c] * (1.f / NTOT) - m * m;
    float y = (xs[idx] - m) * rsqrtf(var + 1e-5f) * g[c] + b[c];
    xs[idx] = y;
    xsb[idx] = (__bf16)y;
  } else {
    int i = idx - NTOT * 32;
    int c = (i % 24) / 3;
    float fn = fnsum[c] * (1.f / (3.f * NTOT));
    xv[i] = xv[i] * vg[c] * rsqrtf(fn + 1e-5f);
  }
}

// ------------------------------------------------------------- a = xs @ eu_lin
__global__ __launch_bounds__(256) void k_lin_a(const __bf16* __restrict__ xsb,
    const __bf16* __restrict__ lint, __bf16* __restrict__ ab) {
  __shared__ __align__(16) __bf16 sA[16 * 32];
  int tid = threadIdx.x, lane = tid & 31, wv = tid >> 5;
  int n0 = blockIdx.x * 16;
  if (tid < 64) ((uint4*)sA)[tid] = ((const uint4*)(xsb + (size_t)n0 * 32))[tid];
  __syncthreads();
  v8f acc = wmma_bf(frag_a(sA, 32, lane),
                    frag_bt(lint + (size_t)(wv * 16) * 32, 32, lane), vzero());
  int nl = (lane & 15) + wv * 16, mbv = (lane >> 4) * 8;
  #pragma unroll
  for (int r = 0; r < 8; ++r)
    ab[(size_t)(n0 + mbv + r) * 128 + nl] = (__bf16)acc[r];
}

// ------------------------------------------------------------- edge update
// 32 edges/block, dual M-tiles per wave; GEMM1 pipelines the B stream.
__global__ __launch_bounds__(256) void k_edge_upd(const __bf16* __restrict__ ab,
    __bf16* __restrict__ efb, float* __restrict__ ef, const int* __restrict__ eidx,
    const __bf16* __restrict__ w1t, const float* __restrict__ b1,
    const __bf16* __restrict__ w2t, const float* __restrict__ b2,
    const __bf16* __restrict__ w3t, const float* __restrict__ b3,
    const float* __restrict__ lng, const float* __restrict__ lnb) {
  __shared__ __align__(16) __bf16 sEin[32 * 384];
  __shared__ __align__(16) __bf16 sU1[32 * 128];
  __shared__ __align__(16) __bf16 sU2[32 * 128];
  __shared__ float sF[32 * 128];
  __shared__ float sMV[32 * 2];
  int tid = threadIdx.x, lane = tid & 31, wv = tid >> 5;
  int e0 = blockIdx.x * 32;
  int nl = (lane & 15) + wv * 16, mb = (lane >> 4) * 8;
  {
    int r = tid >> 4, q = (tid & 15) * 8;
    #pragma unroll
    for (int h = 0; h < 2; ++h) {
      int rr = r + h * 16;
      int d = eidx[e0 + rr], s = eidx[ETOT + e0 + rr];
      *(uint4*)(sEin + rr * 384 + q)       = *(const uint4*)(ab + (size_t)d * 128 + q);
      *(uint4*)(sEin + rr * 384 + 128 + q) = *(const uint4*)(ab + (size_t)s * 128 + q);
      *(uint4*)(sEin + rr * 384 + 256 + q) = *(const uint4*)(efb + (size_t)(e0 + rr) * 128 + q);
    }
  }
  __syncthreads();
  // GEMM1: [32,384]x[384,128] with one-step B prefetch
  v8f acc1a = vzero(), acc1b = vzero();
  const __bf16* bt1 = w1t + (size_t)(wv * 16) * 384;
  v16bf b = frag_bt(bt1, 384, lane);
  #pragma unroll
  for (int ks = 0; ks < 12; ++ks) {
    int kn = (ks < 11) ? (ks + 1) * 32 : ks * 32;
    v16bf bn = frag_bt(bt1 + kn, 384, lane);
    acc1a = wmma_bf(frag_a(sEin + ks * 32, 384, lane), b, acc1a);
    acc1b = wmma_bf(frag_a(sEin + 16 * 384 + ks * 32, 384, lane), b, acc1b);
    b = bn;
  }
  {
    float bb = b1[nl];
    #pragma unroll
    for (int r = 0; r < 8; ++r) {
      float va = acc1a[r] + bb, vb = acc1b[r] + bb;
      sU1[(mb + r) * 128 + nl]      = (__bf16)(va > 0.f ? va : 0.f);
      sU1[(16 + mb + r) * 128 + nl] = (__bf16)(vb > 0.f ? vb : 0.f);
    }
  }
  __syncthreads();
  v8f acc2a = vzero(), acc2b = vzero();
  const __bf16* bt2 = w2t + (size_t)(wv * 16) * 128;
  #pragma unroll
  for (int ks = 0; ks < 4; ++ks) {
    v16bf bf = frag_bt(bt2 + ks * 32, 128, lane);
    acc2a = wmma_bf(frag_a(sU1 + ks * 32, 128, lane), bf, acc2a);
    acc2b = wmma_bf(frag_a(sU1 + 16 * 128 + ks * 32, 128, lane), bf, acc2b);
  }
  {
    float bb = b2[nl];
    #pragma unroll
    for (int r = 0; r < 8; ++r) {
      float va = acc2a[r] + bb, vb = acc2b[r] + bb;
      sU2[(mb + r) * 128 + nl]      = (__bf16)(va > 0.f ? va : 0.f);
      sU2[(16 + mb + r) * 128 + nl] = (__bf16)(vb > 0.f ? vb : 0.f);
    }
  }
  __syncthreads();
  v8f acc3a = vzero(), acc3b = vzero();
  const __bf16* bt3 = w3t + (size_t)(wv * 16) * 128;
  #pragma unroll
  for (int ks = 0; ks < 4; ++ks) {
    v16bf bf = frag_bt(bt3 + ks * 32, 128, lane);
    acc3a = wmma_bf(frag_a(sU2 + ks * 32, 128, lane), bf, acc3a);
    acc3b = wmma_bf(frag_a(sU2 + 16 * 128 + ks * 32, 128, lane), bf, acc3b);
  }
  {
    float bb = b3[nl];
    #pragma unroll
    for (int r = 0; r < 8; ++r) {
      sF[(mb + r) * 128 + nl]      = acc3a[r] + bb + ef[(size_t)(e0 + mb + r) * 128 + nl];
      sF[(16 + mb + r) * 128 + nl] = acc3b[r] + bb + ef[(size_t)(e0 + 16 + mb + r) * 128 + nl];
    }
  }
  __syncthreads();
  if (tid < 32) {
    float m = 0.f;
    for (int c = 0; c < 128; ++c) m += sF[tid * 128 + c];
    m *= (1.f / 128.f);
    float v = 0.f;
    for (int c = 0; c < 128; ++c) { float d = sF[tid * 128 + c] - m; v += d * d; }
    sMV[tid * 2] = m;
    sMV[tid * 2 + 1] = rsqrtf(v * (1.f / 128.f) + 1e-5f);
  }
  __syncthreads();
  #pragma unroll
  for (int it = 0; it < 16; ++it) {
    int i = it * 256 + tid;
    int r = i >> 7, c = i & 127;
    float y = (sF[i] - sMV[r * 2]) * sMV[r * 2 + 1] * lng[c] + lnb[c];
    ef[(size_t)(e0 + r) * 128 + c]  = y;
    efb[(size_t)(e0 + r) * 128 + c] = (__bf16)y;
  }
}

// ------------------------------------------------------------- readout
__global__ __launch_bounds__(256) void k_readout(const float* __restrict__ xs,
    const float* __restrict__ xv, const float* __restrict__ rot,
    const __bf16* __restrict__ mulvt, const float* __restrict__ mub,
    const float* __restrict__ lvb, float* __restrict__ out) {
  __shared__ __align__(16) __bf16 sFt[16 * 64];
  int tid = threadIdx.x, lane = tid & 31, wv = tid >> 5;
  int n0 = blockIdx.x * 16;
  #pragma unroll
  for (int it = 0; it < 4; ++it) {
    int i = it * 256 + tid;
    int r = i >> 6, c = i & 63;
    float v = 0.f;
    if (c < 32) v = xs[(size_t)(n0 + r) * 32 + c];
    else if (c < 56) {
      int q = c - 32, ch = q / 3, y = q % 3;
      const float* R = rot + (size_t)(n0 + r) * 9;
      const float* V = xv + (size_t)(n0 + r) * 24 + ch * 3;
      v = R[y] * V[0] + R[3 + y] * V[1] + R[6 + y] * V[2];
    }
    sFt[i] = (__bf16)v;
  }
  __syncthreads();
  v8f a0 = vzero(), a1 = vzero();
  #pragma unroll
  for (int ks = 0; ks < 2; ++ks) {
    int kk = ks * 32;
    a0 = wmma_bf(frag_a(sFt + kk, 64, lane),
                 frag_bt(mulvt + (size_t)(wv * 16) * 64 + kk, 64, lane), a0);
    a1 = wmma_bf(frag_a(sFt + kk, 64, lane),
                 frag_bt(mulvt + (size_t)((wv + 8) * 16) * 64 + kk, 64, lane), a1);
  }
  int mbv = (lane >> 4) * 8;
  #pragma unroll
  for (int t = 0; t < 2; ++t) {
    int nc = (lane & 15) + (wv + t * 8) * 16;
    float bb = (nc < 128) ? mub[nc] : lvb[nc - 128];
    v8f a = t ? a1 : a0;
    #pragma unroll
    for (int r = 0; r < 8; ++r) {
      int node = n0 + mbv + r;
      float y = a[r] + bb;
      if (nc < 128) out[(size_t)node * 128 + nc] = y;
      else out[(size_t)NTOT * 128 + (size_t)node * 128 + (nc - 128)] = y;
    }
  }
}

// ---------------------------------------------------------------- launcher
extern "C" void kernel_launch(void* const* d_in, const int* in_sizes, int n_in,
                              void* d_out, int out_size, void* d_ws, size_t ws_size,
                              hipStream_t stream) {
  (void)in_sizes; (void)n_in; (void)out_size; (void)ws_size;
  const float* node_raw  = (const float*)d_in[0];
  const float* edge_raw  = (const float*)d_in[1];
  const float* edge_vecs = (const float*)d_in[2];
  const float* rot       = (const float*)d_in[3];
  const int*   eidx      = (const int*)d_in[4];
  const float* ee_w1 = (const float*)d_in[5],  *ee_b1 = (const float*)d_in[6];
  const float* ee_w2 = (const float*)d_in[7],  *ee_b2 = (const float*)d_in[8];
  const float* ee_w3 = (const float*)d_in[9],  *ee_b3 = (const float*)d_in[10];
  const float* ee_lng = (const float*)d_in[11], *ee_lnb = (const float*)d_in[12];
  const float* ne_ws = (const float*)d_in[13], *ne_wv = (const float*)d_in[14];
  const float* fc_w1 = (const float*)d_in[15], *fc_b1 = (const float*)d_in[16];
  const float* fc_w2 = (const float*)d_in[17], *fc_b2 = (const float*)d_in[18];
  const float* bn_g = (const float*)d_in[19], *bn_b = (const float*)d_in[20];
  const float* bn_vg = (const float*)d_in[21];
  const float* eu_lin = (const float*)d_in[22];
  const float* eu_w1 = (const float*)d_in[23], *eu_b1 = (const float*)d_in[24];
  const float* eu_w2 = (const float*)d_in[25], *eu_b2 = (const float*)d_in[26];
  const float* eu_w3 = (const float*)d_in[27], *eu_b3 = (const float*)d_in[28];
  const float* eu_lng = (const float*)d_in[29], *eu_lnb = (const float*)d_in[30];
  const float* mu_w = (const float*)d_in[31], *mu_b = (const float*)d_in[32];
  const float* lv_w = (const float*)d_in[33], *lv_b = (const float*)d_in[34];

  char* ws = (char*)d_ws;
  float*  ef   = (float*)ws;                       // E*128 f32
  __bf16* efb  = (__bf16*)(ws + 15728640);         // E*128 bf16
  float*  xs   = (float*)(ws + 23592960);          // N*32
  float*  xv   = (float*)(ws + 23724032);          // N*24
  __bf16* xsb  = (__bf16*)(ws + 23822336);         // N*32 bf16
  __bf16* ab   = (__bf16*)(ws + 23887872);         // N*128 bf16
  float*  cnt  = (float*)(ws + 24150016);          // N
  float*  ags  = (float*)(ws + 24154112);          // N*32
  float*  agv  = (float*)(ws + 24285184);          // N*24
  float*  bnsum = (float*)(ws + 24383488);         // 32
  float*  bnsq  = bnsum + 32;
  float*  fnsum = bnsq + 32;
  // transposed bf16 weights (element offsets from wt)
  __bf16* wt = (__bf16*)(ws + 24384000);
  __bf16* ee_w1_t  = wt;                  // [128][1664]
  __bf16* ee_w2_t  = wt + 212992;         // [128][128]
  __bf16* ee_w3_t  = wt + 229376;
  __bf16* fc_w1_t  = wt + 245760;         // 4 x [128][128]
  __bf16* fc_w2_t  = wt + 311296;         // 4 x [1664][128]
  __bf16* eu_lin_t = wt + 1163264;        // 4 x [128][32]
  __bf16* eu_w1_t  = wt + 1179648;        // 4 x [128][384]
  __bf16* eu_w2_t  = wt + 1376256;        // 4 x [128][128]
  __bf16* eu_w3_t  = wt + 1441792;
  __bf16* mulv_t   = wt + 1507328;        // [256][64] (mu|lv, K padded 56->64)

  auto T = [&](const float* s, __bf16* d, int K, int N, int Kpad) {
    k_transpose<<<(N * Kpad + 255) / 256, 256, 0, stream>>>(s, d, K, N, Kpad);
  };
  T(ee_w1, ee_w1_t, 1664, 128, 1664);
  T(ee_w2, ee_w2_t, 128, 128, 128);
  T(ee_w3, ee_w3_t, 128, 128, 128);
  for (int l = 0; l < 4; ++l) {
    T(fc_w1 + (size_t)l * 16384, fc_w1_t + (size_t)l * 16384, 128, 128, 128);
    T(fc_w2 + (size_t)l * 212992, fc_w2_t + (size_t)l * 212992, 128, 1664, 128);
    T(eu_lin + (size_t)l * 4096, eu_lin_t + (size_t)l * 4096, 32, 128, 32);
    T(eu_w1 + (size_t)l * 49152, eu_w1_t + (size_t)l * 49152, 384, 128, 384);
    T(eu_w2 + (size_t)l * 16384, eu_w2_t + (size_t)l * 16384, 128, 128, 128);
    T(eu_w3 + (size_t)l * 16384, eu_w3_t + (size_t)l * 16384, 128, 128, 128);
  }
  T(mu_w, mulv_t, 56, 128, 64);
  T(lv_w, mulv_t + 8192, 56, 128, 64);

  k_zero<<<4, 256, 0, stream>>>(cnt, NTOT);
  k_cnt<<<ETOT / 256, 256, 0, stream>>>(eidx, cnt);
  k_node_embed<<<(NTOT * 56 + 255) / 256, 256, 0, stream>>>(node_raw, ne_ws, ne_wv, xs, xv);
  k_edge_embed<<<ETOT / 32, 256, K1_SMEM, stream>>>(edge_raw, ee_w1_t, ee_b1, ee_w2_t, ee_b2,
                                                    ee_w3_t, ee_b3, ee_lng, ee_lnb, ef, efb);
  for (int l = 0; l < 4; ++l) {
    k_zero<<<(NTOT * 56 + 72 + 255) / 256, 256, 0, stream>>>(ags, NTOT * 32 + NTOT * 24 + 72);
    k_fc_tp<<<ETOT / 32, 256, K2_SMEM, stream>>>(efb, xs, xv, edge_vecs, eidx,
        fc_w1_t + (size_t)l * 16384, fc_b1 + (size_t)l * 128,
        fc_w2_t + (size_t)l * 212992, fc_b2 + (size_t)l * 1664, ags, agv);
    k_node_s<<<NTOT * 32 / 256, 256, 0, stream>>>(xs, ags, cnt, bnsum, bnsq);
    k_node_v<<<NTOT * 8 / 256, 256, 0, stream>>>(xv, agv, cnt, fnsum);
    k_bn_apply<<<(NTOT * 56) / 256, 256, 0, stream>>>(xs, xv, xsb,
        bn_g + l * 32, bn_b + l * 32, bn_vg + l * 8, bnsum, bnsq, fnsum);
    k_lin_a<<<NTOT / 16, 256, 0, stream>>>(xsb, eu_lin_t + (size_t)l * 4096, ab);
    k_edge_upd<<<ETOT / 32, 256, 0, stream>>>(ab, efb, ef, eidx,
        eu_w1_t + (size_t)l * 49152, eu_b1 + (size_t)l * 128,
        eu_w2_t + (size_t)l * 16384, eu_b2 + (size_t)l * 128,
        eu_w3_t + (size_t)l * 16384, eu_b3 + (size_t)l * 128,
        eu_lng + (size_t)l * 128, eu_lnb + (size_t)l * 128);
  }
  k_readout<<<NTOT / 16, 256, 0, stream>>>(xs, xv, rot, mulv_t, mu_b, lv_b, (float*)d_out);
}
